// AenetMACE_19739669692989
// MI455X (gfx1250) — matrix operand order repaired and checked by reading, compiled, not run on hardware
//
#include <hip/hip_runtime.h>
#include <hip/hip_bf16.h>
#include <math.h>

// ---------------------------------------------------------------------------
// Types for CDNA5 WMMA (wave32, 16x16x32 bf16 -> f32)
// ---------------------------------------------------------------------------
typedef __attribute__((ext_vector_type(16))) __bf16 bf16x16;
typedef __attribute__((ext_vector_type(8)))  float  f32x8;

#define WMMA_BF16(a, b, c) \
  __builtin_amdgcn_wmma_f32_16x16x32_bf16(false, (a), false, (b), (short)0, (c), false, false)

#define C_DIM 64
#define Z_DIM 10

__device__ __forceinline__ float silu_f(float x) { return x / (1.f + __expf(-x)); }

__device__ __forceinline__ f32x8 zero8() {
  f32x8 z;
#pragma unroll
  for (int i = 0; i < 8; ++i) z[i] = 0.f;
  return z;
}

__device__ __forceinline__ void atom_add_f32(float* p, float v) {
  unsafeAtomicAdd(p, v);  // lowers to global_atomic_add_f32 on gfx1250
}

// ---------------------------------------------------------------------------
// A-fragment loader: X row-major f32, row stride 64, no bounds checks.
// Lane L holds row M = L&15; K set = kb + koff + {0..15}, koff = 16*(L>=16).
// One contiguous 16-float run -> 4x global_load_b128.
// ---------------------------------------------------------------------------
__device__ __forceinline__ bf16x16 load_A64(const float* __restrict__ X,
                                            int row0, int kb, int lane) {
  const float* rp = X + (size_t)(row0 + (lane & 15)) * 64 + kb + ((lane >> 4) << 4);
  float4 x0 = *(const float4*)(rp);
  float4 x1 = *(const float4*)(rp + 4);
  float4 x2 = *(const float4*)(rp + 8);
  float4 x3 = *(const float4*)(rp + 12);
  bf16x16 a;
  a[0] = (__bf16)x0.x;  a[1] = (__bf16)x0.y;  a[2]  = (__bf16)x0.z;  a[3]  = (__bf16)x0.w;
  a[4] = (__bf16)x1.x;  a[5] = (__bf16)x1.y;  a[6]  = (__bf16)x1.z;  a[7]  = (__bf16)x1.w;
  a[8] = (__bf16)x2.x;  a[9] = (__bf16)x2.y;  a[10] = (__bf16)x2.z;  a[11] = (__bf16)x2.w;
  a[12] = (__bf16)x3.x; a[13] = (__bf16)x3.y; a[14] = (__bf16)x3.z;  a[15] = (__bf16)x3.w;
  return a;
}

// ---------------------------------------------------------------------------
// B-fragment loader from pre-transposed bf16 weights WT[Nout][Kpad].
// Lane L holds col N = n0 + (L&15); element e -> K = kb + 16*(L>=16) + e.
// 16 contiguous bf16 per lane -> 2x global_load_b128.
// ---------------------------------------------------------------------------
__device__ __forceinline__ bf16x16 load_Bt(const __bf16* __restrict__ WT, int Kpad,
                                           int kb, int n0, int lane) {
  const __bf16* wp = WT + (size_t)(n0 + (lane & 15)) * Kpad + kb + ((lane >> 4) << 4);
  return *(const bf16x16*)wp;
}

// ---------------------------------------------------------------------------
// Weight prep: W[K x Nout] f32 row-major -> WT[Nout x Kpad] bf16 (zero padded)
// ---------------------------------------------------------------------------
__global__ __launch_bounds__(256) void prep_weight(const float* __restrict__ W,
                                                   __bf16* __restrict__ WT,
                                                   int K, int Nout, int Kpad) {
  int i = blockIdx.x * 256 + threadIdx.x;
  if (i >= Nout * Kpad) return;
  int n = i / Kpad, k = i - n * Kpad;
  WT[i] = (k < K) ? (__bf16)W[(size_t)k * Nout + n] : (__bf16)0.f;
}

// ---------------------------------------------------------------------------
// GEMM: Y[R x 64] = act( X[R x 64] @ W[64 x 64] ), bf16 WMMA, K = 64.
// ---------------------------------------------------------------------------
template <bool SILU>
__global__ __launch_bounds__(256) void gemm64(const float* __restrict__ X,
                                              const __bf16* __restrict__ WT,
                                              float* __restrict__ Y, int R) {
  int lane = threadIdx.x & 31;
  int tile = blockIdx.x * 8 + (threadIdx.x >> 5);
  int row0 = tile * 16;
  if (row0 >= R) return;  // wave-uniform exit: EXEC all-ones for WMMA

  f32x8 acc[4];
#pragma unroll
  for (int t = 0; t < 4; ++t) acc[t] = zero8();

#pragma unroll
  for (int kb = 0; kb < 64; kb += 32) {
    bf16x16 a = load_A64(X, row0, kb, lane);
#pragma unroll
    for (int t = 0; t < 4; ++t)
      acc[t] = WMMA_BF16(a, load_Bt(WT, 64, kb, t * 16, lane), acc[t]);
  }

  int mrow = row0 + ((lane >> 4) << 3);
  int ncol = lane & 15;
  if (row0 + 16 <= R) {
#pragma unroll
    for (int t = 0; t < 4; ++t)
#pragma unroll
      for (int r = 0; r < 8; ++r) {
        float x = acc[t][r];
        if (SILU) x = silu_f(x);
        Y[(size_t)(mrow + r) * 64 + t * 16 + ncol] = x;
      }
  } else {
#pragma unroll
    for (int t = 0; t < 4; ++t)
#pragma unroll
      for (int r = 0; r < 8; ++r) {
        float x = acc[t][r];
        if (SILU) x = silu_f(x);
        if (mrow + r < R) Y[(size_t)(mrow + r) * 64 + t * 16 + ncol] = x;
      }
  }
}

// ---------------------------------------------------------------------------
// radial layer 1 (K = 8, pure VALU): h1 = silu(efeat @ RW1), RW1 in LDS.
// ---------------------------------------------------------------------------
__global__ __launch_bounds__(256) void radial1(const float* __restrict__ efeat,
                                               const float* __restrict__ RW1,
                                               float* __restrict__ h1, int E) {
  __shared__ float wsm[8 * 64];
  int t = threadIdx.x;
  wsm[t] = RW1[t];
  wsm[t + 256] = RW1[t + 256];
  __syncthreads();
  int c = t & 63;
  int e = blockIdx.x * 4 + (t >> 6);
  if (e >= E) return;
  const float* ef = efeat + (size_t)e * 8;
  float4 f0 = *(const float4*)ef;
  float4 f1 = *(const float4*)(ef + 4);
  float acc = f0.x * wsm[c]       + f0.y * wsm[64 + c]
            + f0.z * wsm[128 + c] + f0.w * wsm[192 + c]
            + f1.x * wsm[256 + c] + f1.y * wsm[320 + c]
            + f1.z * wsm[384 + c] + f1.w * wsm[448 + c];
  h1[(size_t)e * 64 + c] = silu_f(acc);
}

// ---------------------------------------------------------------------------
// node_init: species = argmax(one-hot), s = W_embed[species]
// ---------------------------------------------------------------------------
__global__ __launch_bounds__(128) void node_init(const float* __restrict__ attrs,
                                                 const float* __restrict__ Wemb,
                                                 int* __restrict__ species,
                                                 float* __restrict__ s, int N) {
  int n = blockIdx.x * blockDim.x + threadIdx.x;
  if (n >= N) return;
  int best = 0;
  float bv = attrs[(size_t)n * Z_DIM];
#pragma unroll
  for (int z = 1; z < Z_DIM; ++z) {
    float a = attrs[(size_t)n * Z_DIM + z];
    if (a > bv) { bv = a; best = z; }
  }
  species[n] = best;
  const float4* wr = (const float4*)(Wemb + (size_t)best * C_DIM);
  float4* sr = (float4*)(s + (size_t)n * C_DIM);
#pragma unroll
  for (int j = 0; j < 16; ++j) sr[j] = wr[j];
}

// ---------------------------------------------------------------------------
// species sort: histogram -> prefix -> scatter permutation
// ---------------------------------------------------------------------------
__global__ __launch_bounds__(64) void hist_zero(int* __restrict__ hist) {
  if (threadIdx.x < 48) hist[threadIdx.x] = 0;
}
__global__ __launch_bounds__(256) void hist_count(const int* __restrict__ species,
                                                  int* __restrict__ hist, int N) {
  int n = blockIdx.x * 256 + threadIdx.x;
  if (n < N) atomicAdd(&hist[species[n]], 1);
}
__global__ __launch_bounds__(32) void hist_prefix(int* __restrict__ hist) {
  // hist[0..9] counts; hist[16..25] = exclusive offsets; hist[32..41] = cursors
  if (threadIdx.x == 0 && blockIdx.x == 0) {
    int run = 0;
    for (int z = 0; z < Z_DIM; ++z) {
      hist[16 + z] = run;
      hist[32 + z] = run;
      run += hist[z];
    }
  }
}
__global__ __launch_bounds__(256) void perm_scatter(const int* __restrict__ species,
                                                    int* __restrict__ hist,
                                                    int* __restrict__ perm, int N) {
  int n = blockIdx.x * 256 + threadIdx.x;
  if (n >= N) return;
  int pos = atomicAdd(&hist[32 + species[n]], 1);
  perm[pos] = n;
}

// ---------------------------------------------------------------------------
// edge_geom
// ---------------------------------------------------------------------------
__global__ __launch_bounds__(128) void edge_geom(const float* __restrict__ pos,
                                                 const float* __restrict__ shifts,
                                                 const int* __restrict__ eidx,
                                                 float* __restrict__ Y1,
                                                 float* __restrict__ efeat, int E) {
  int e = blockIdx.x * blockDim.x + threadIdx.x;
  if (e >= E) return;
  int snd = eidx[e], rcv = eidx[E + e];
  float dx = pos[(size_t)rcv * 3 + 0] - pos[(size_t)snd * 3 + 0] + shifts[(size_t)e * 3 + 0];
  float dy = pos[(size_t)rcv * 3 + 1] - pos[(size_t)snd * 3 + 1] + shifts[(size_t)e * 3 + 1];
  float dz = pos[(size_t)rcv * 3 + 2] - pos[(size_t)snd * 3 + 2] + shifts[(size_t)e * 3 + 2];
  float r = sqrtf(dx * dx + dy * dy + dz * dz);
  float inv = 1.f / (r + 1e-9f);
  const float SQRT3 = 1.7320508075688772f;
  Y1[(size_t)e * 3 + 0] = SQRT3 * dx * inv;
  Y1[(size_t)e * 3 + 1] = SQRT3 * dy * inv;
  Y1[(size_t)e * 3 + 2] = SQRT3 * dz * inv;
  float u = r * 0.2f;
  float env = 0.f;
  if (u < 1.f) {
    float u2 = u * u, u6 = u2 * u2 * u2;
    env = 1.f - 28.f * u6 + 48.f * u6 * u - 21.f * u6 * u2;
  }
  const float PREF = 0.6324555320336759f;
  const float PI = 3.14159265358979f;
  float scale = PREF * inv * env;
#pragma unroll
  for (int n = 1; n <= 8; ++n)
    efeat[(size_t)e * 8 + (n - 1)] = scale * __sinf((float)n * PI * u);
}

// ---------------------------------------------------------------------------
// zero fill
// ---------------------------------------------------------------------------
__global__ __launch_bounds__(256) void zero_f32(float* __restrict__ p, size_t n) {
  size_t i = (size_t)blockIdx.x * blockDim.x + threadIdx.x;
  size_t stride = (size_t)gridDim.x * blockDim.x;
  for (; i < n; i += stride) p[i] = 0.f;
}

// ---------------------------------------------------------------------------
// edge_msg: per 16-edge tile (one wave):
//   phase 1: w[16 x 320] = h2[16 x 64] @ RW3[64 x 320] via 40 WMMAs -> LDS
//   phase 2: gather sender feats, build m0/m1, atomic-scatter to receiver aggs
// ---------------------------------------------------------------------------
__global__ __launch_bounds__(32) void edge_msg(const float* __restrict__ h2,
                                               const __bf16* __restrict__ RW3T,
                                               const float* __restrict__ s_up,
                                               const float* __restrict__ v_up,
                                               const float* __restrict__ Y1,
                                               const int* __restrict__ eidx,
                                               float* __restrict__ agg0,
                                               float* __restrict__ agg1,
                                               int E, size_t NC) {
  __shared__ float wbuf[16 * 320];
  __shared__ float ybuf[16 * 3];
  int lane = threadIdx.x;
  int e0 = blockIdx.x * 16;
  if (e0 >= E) return;

  bf16x16 a0 = load_A64(h2, e0, 0, lane);
  bf16x16 a1 = load_A64(h2, e0, 32, lane);
  int mrow = (lane >> 4) << 3;
  int ncol = lane & 15;
#pragma unroll
  for (int t = 0; t < 20; ++t) {
    f32x8 acc = zero8();
    acc = WMMA_BF16(a0, load_Bt(RW3T, 64, 0, t * 16, lane), acc);
    acc = WMMA_BF16(a1, load_Bt(RW3T, 64, 32, t * 16, lane), acc);
#pragma unroll
    for (int r = 0; r < 8; ++r)
      wbuf[(mrow + r) * 320 + t * 16 + ncol] = acc[r];
  }
  if (lane < 16) {
    int e = e0 + lane;
    if (e < E) {
      ybuf[lane * 3 + 0] = Y1[(size_t)e * 3 + 0];
      ybuf[lane * 3 + 1] = Y1[(size_t)e * 3 + 1];
      ybuf[lane * 3 + 2] = Y1[(size_t)e * 3 + 2];
    }
  }
  __syncthreads();

  const float INV_S3 = 0.5773502691896258f;
  const float INV_S2 = 0.7071067811865476f;
  const float INV_AVG = 1.f / 32.f;
#pragma unroll 1
  for (int it = 0; it < 32; ++it) {
    int idx = it * 32 + lane;       // 16 edges x 64 channels
    int el = idx >> 6, c = idx & 63;
    int e = e0 + el;
    if (e >= E) continue;
    int snd = eidx[e], rcv = eidx[E + e];
    size_t sb = (size_t)snd * 64 + c;
    float sj = s_up[sb];
    float vj0 = v_up[0 * NC + sb], vj1 = v_up[1 * NC + sb], vj2 = v_up[2 * NC + sb];
    float y0 = ybuf[el * 3 + 0], y1 = ybuf[el * 3 + 1], y2 = ybuf[el * 3 + 2];
    const float* wr = &wbuf[el * 320];
    float w00 = wr[c], w110 = wr[64 + c], w011 = wr[128 + c], w101 = wr[192 + c], w111 = wr[256 + c];
    float dotvy = vj0 * y0 + vj1 * y1 + vj2 * y2;
    float m0 = w00 * sj + w110 * dotvy * INV_S3;
    float cx = vj1 * y2 - vj2 * y1;
    float cy = vj2 * y0 - vj0 * y2;
    float cz = vj0 * y1 - vj1 * y0;
    float m1x = w011 * sj * y0 + w101 * vj0 + w111 * cx * INV_S2;
    float m1y = w011 * sj * y1 + w101 * vj1 + w111 * cy * INV_S2;
    float m1z = w011 * sj * y2 + w101 * vj2 + w111 * cz * INV_S2;
    size_t rb = (size_t)rcv * 64 + c;
    atom_add_f32(&agg0[rb], m0 * INV_AVG);
    atom_add_f32(&agg1[0 * NC + rb], m1x * INV_AVG);
    atom_add_f32(&agg1[1 * NC + rb], m1y * INV_AVG);
    atom_add_f32(&agg1[2 * NC + rb], m1z * INV_AVG);
  }
}

// ---------------------------------------------------------------------------
// prod_build (elementwise): prodS/prodV from ms/mv and species polys P0/P1
// ---------------------------------------------------------------------------
__global__ __launch_bounds__(256) void prod_build(const int* __restrict__ species,
                                                  const float* __restrict__ ms,
                                                  const float* __restrict__ mv,
                                                  const float* __restrict__ P0,
                                                  const float* __restrict__ P1,
                                                  float* __restrict__ prodS,
                                                  float* __restrict__ prodV,
                                                  int N, size_t NC) {
  int i = blockIdx.x * 256 + threadIdx.x;
  if (i >= N * C_DIM) return;
  int n = i >> 6, c = i & 63;
  int z = species[n];
  const float* p = P0 + ((size_t)z * 64 + c) * 3;
  const float* q = P1 + ((size_t)z * 64 + c) * 2;
  float msc = ms[i];
  float mv0 = mv[0 * NC + i], mv1 = mv[1 * NC + i], mv2 = mv[2 * NC + i];
  float nv2 = mv0 * mv0 + mv1 * mv1 + mv2 * mv2;
  prodS[i] = p[0] * msc + p[1] * msc * msc + p[2] * nv2;
  float qf = q[0] + q[1] * msc;
  prodV[0 * NC + i] = qf * mv0;
  prodV[1 * NC + i] = qf * mv1;
  prodV[2 * NC + i] = qf * mv2;
}

// ---------------------------------------------------------------------------
// sc_update (species-sorted): blocks process 8 same-species nodes with
// Wsc_s[z]/Wsc_v[z] staged once in LDS (32 KB). Adds the species matvec
// to the WMMA prod-GEMM result and writes new s, v, and the descriptor.
// grid = (ceil(N/8), Z_DIM); blocks beyond the species count exit early.
// ---------------------------------------------------------------------------
__global__ __launch_bounds__(64) void sc_update(const int* __restrict__ hist,
                                                const int* __restrict__ perm,
                                                float* __restrict__ s, float* __restrict__ v,
                                                const float* __restrict__ gs,   // prodS @ Wprod_s
                                                const float* __restrict__ gv,   // prodV @ Wprod_v
                                                const float* __restrict__ Wsc_s,
                                                const float* __restrict__ Wsc_v,
                                                float* __restrict__ out,
                                                int N, size_t NC, int layer, int Ltot) {
  __shared__ float Wls[64 * 64];
  __shared__ float Wlv[64 * 64];
  __shared__ float sbuf[64], vbuf[3][64];
  int z = blockIdx.y;
  int cnt = hist[z];
  int start = hist[16 + z];
  int j0 = blockIdx.x * 8;
  if (j0 >= cnt) return;
  int c = threadIdx.x;

  const float* Ws = Wsc_s + (size_t)z * 4096;
  const float* Wv = Wsc_v + (size_t)z * 4096;
#pragma unroll 8
  for (int t = c; t < 4096; t += 64) {
    Wls[t] = Ws[t];
    Wlv[t] = Wv[t];
  }
  __syncthreads();

  for (int jj = 0; jj < 8; ++jj) {
    int j = j0 + jj;
    if (j >= cnt) break;                 // block-uniform
    int n = perm[start + j];
    size_t base = (size_t)n * 64 + c;
    sbuf[c] = s[base];
    vbuf[0][c] = v[0 * NC + base];
    vbuf[1][c] = v[1 * NC + base];
    vbuf[2][c] = v[2 * NC + base];
    __syncthreads();
    float scs = 0.f, scv0 = 0.f, scv1 = 0.f, scv2 = 0.f;
#pragma unroll 4
    for (int k = 0; k < 64; ++k) {
      float ws = Wls[k * 64 + c], wv = Wlv[k * 64 + c];
      scs  += sbuf[k] * ws;
      scv0 += vbuf[0][k] * wv;
      scv1 += vbuf[1][k] * wv;
      scv2 += vbuf[2][k] * wv;
    }
    float snew = gs[base] + scs;
    s[base] = snew;
    v[0 * NC + base] = gv[0 * NC + base] + scv0;
    v[1 * NC + base] = gv[1 * NC + base] + scv1;
    v[2 * NC + base] = gv[2 * NC + base] + scv2;
    out[(size_t)n * ((size_t)Ltot * 64) + (size_t)layer * 64 + c] = snew;
    __syncthreads();
  }
}

// ---------------------------------------------------------------------------
// Host launch
// ---------------------------------------------------------------------------
extern "C" void kernel_launch(void* const* d_in, const int* in_sizes, int n_in,
                              void* d_out, int out_size, void* d_ws, size_t ws_size,
                              hipStream_t stream) {
  const float* node_attrs = (const float*)d_in[0];
  const float* pos        = (const float*)d_in[1];
  const float* shifts     = (const float*)d_in[2];
  const float* Wemb       = (const float*)d_in[3];
  const float* Wup_s      = (const float*)d_in[4];
  const float* Wup_v      = (const float*)d_in[5];
  const float* RW1        = (const float*)d_in[6];
  const float* RW2        = (const float*)d_in[7];
  const float* RW3        = (const float*)d_in[8];
  const float* Wout_s     = (const float*)d_in[9];
  const float* Wout_v     = (const float*)d_in[10];
  const float* Wsc_s      = (const float*)d_in[11];
  const float* Wsc_v      = (const float*)d_in[12];
  const float* P0         = (const float*)d_in[13];
  const float* P1         = (const float*)d_in[14];
  const float* Wprod_s    = (const float*)d_in[15];
  const float* Wprod_v    = (const float*)d_in[16];
  const int*   eidx       = (const int*)d_in[17];

  const int N = in_sizes[0] / Z_DIM;
  const int E = in_sizes[17] / 2;
  const int L = in_sizes[4] / (C_DIM * C_DIM);
  const size_t NC = (size_t)N * C_DIM;

  // workspace layout (f32 region, then ints, then bf16 weight cache)
  float* w = (float*)d_ws;
  float* s_    = w; w += NC;
  float* v_    = w; w += 3 * NC;
  float* s_up  = w; w += NC;     // reused as prodS after edge_msg
  float* v_up  = w; w += 3 * NC; // reused as prodV after edge_msg
  float* agg0  = w; w += NC;     // agg0/agg1 contiguous -> one zero fill
  float* agg1  = w; w += 3 * NC;
  float* ms_   = w; w += NC;     // reused as prod-GEMM output
  float* mv_   = w; w += 3 * NC;
  float* Y1    = w; w += (size_t)E * 3;
  float* efeat = w; w += (size_t)E * 8;
  float* h1    = w; w += (size_t)E * 64;
  float* h2    = w; w += (size_t)E * 64;
  int* species = (int*)w;
  int* perm    = species + N;
  int* hist    = perm + N;       // [0..9] counts, [16..25] offsets, [32..41] cursors
  uintptr_t up = (uintptr_t)(hist + 48);
  up = (up + 255) & ~(uintptr_t)255;
  __bf16* wtb = (__bf16*)up;
  // per-layer bf16 cache: 7x 64x64 + 1x 320x64 = 49152 elements
  const size_t WPL = 7 * 4096 + 320 * 64;

  const int ntilesN = (N + 15) / 16, gN = (ntilesN + 7) / 8;
  const int ntilesE = (E + 15) / 16, gE = (ntilesE + 7) / 8;

  zero_f32<<<1024, 256, 0, stream>>>(v_, 3 * NC);
  node_init<<<(N + 127) / 128, 128, 0, stream>>>(node_attrs, Wemb, species, s_, N);
  edge_geom<<<(E + 127) / 128, 128, 0, stream>>>(pos, shifts, eidx, Y1, efeat, E);

  // species sort (once; species fixed across layers)
  hist_zero<<<1, 64, 0, stream>>>(hist);
  hist_count<<<(N + 255) / 256, 256, 0, stream>>>(species, hist, N);
  hist_prefix<<<1, 32, 0, stream>>>(hist);
  perm_scatter<<<(N + 255) / 256, 256, 0, stream>>>(species, hist, perm, N);

  // pre-convert WMMA weights to bf16 transposed layout
  for (int i = 0; i < L; ++i) {
    const size_t cc = (size_t)i * 64 * 64;
    __bf16* base = wtb + (size_t)i * WPL;
    prep_weight<<<16, 256, 0, stream>>>(Wup_s + cc, base, 64, 64, 64);
    prep_weight<<<16, 256, 0, stream>>>(Wup_v + cc, base + 4096, 64, 64, 64);
    prep_weight<<<16, 256, 0, stream>>>(RW2 + cc, base + 2 * 4096, 64, 64, 64);
    prep_weight<<<16, 256, 0, stream>>>(Wout_s + cc, base + 3 * 4096, 64, 64, 64);
    prep_weight<<<16, 256, 0, stream>>>(Wout_v + cc, base + 4 * 4096, 64, 64, 64);
    prep_weight<<<16, 256, 0, stream>>>(Wprod_s + cc, base + 5 * 4096, 64, 64, 64);
    prep_weight<<<16, 256, 0, stream>>>(Wprod_v + cc, base + 6 * 4096, 64, 64, 64);
    prep_weight<<<80, 256, 0, stream>>>(RW3 + (size_t)i * 64 * 320, base + 7 * 4096, 64, 320, 64);
  }

  for (int i = 0; i < L; ++i) {
    __bf16* base = wtb + (size_t)i * WPL;
    __bf16* Wup_sT   = base;
    __bf16* Wup_vT   = base + 4096;
    __bf16* RW2T     = base + 2 * 4096;
    __bf16* Wout_sT  = base + 3 * 4096;
    __bf16* Wout_vT  = base + 4 * 4096;
    __bf16* Wprod_sT = base + 5 * 4096;
    __bf16* Wprod_vT = base + 6 * 4096;
    __bf16* RW3T     = base + 7 * 4096;

    // up-projections (WMMA)
    gemm64<false><<<gN, 256, 0, stream>>>(s_, Wup_sT, s_up, N);
    for (int d = 0; d < 3; ++d)
      gemm64<false><<<gN, 256, 0, stream>>>(v_ + d * NC, Wup_vT, v_up + d * NC, N);
    // radial MLP
    radial1<<<(E + 3) / 4, 256, 0, stream>>>(efeat, RW1 + (size_t)i * 8 * 64, h1, E);
    gemm64<true><<<gE, 256, 0, stream>>>(h1, RW2T, h2, E);
    // messages: fused RW3 WMMA + gather/scatter
    zero_f32<<<1024, 256, 0, stream>>>(agg0, 4 * NC);
    edge_msg<<<ntilesE, 32, 0, stream>>>(h2, RW3T, s_up, v_up, Y1, eidx, agg0, agg1, E, NC);
    // out-projections (WMMA)
    gemm64<false><<<gN, 256, 0, stream>>>(agg0, Wout_sT, ms_, N);
    for (int d = 0; d < 3; ++d)
      gemm64<false><<<gN, 256, 0, stream>>>(agg1 + d * NC, Wout_vT, mv_ + d * NC, N);
    // products: elementwise build (into s_up/v_up), then WMMA GEMMs (into ms_/mv_)
    prod_build<<<((int)(NC) + 255) / 256, 256, 0, stream>>>(species, ms_, mv_, 
                                      P0 + (size_t)i * Z_DIM * 64 * 3,
                                      P1 + (size_t)i * Z_DIM * 64 * 2,
                                      s_up, v_up, N, NC);
    gemm64<false><<<gN, 256, 0, stream>>>(s_up, Wprod_sT, ms_, N);
    for (int d = 0; d < 3; ++d)
      gemm64<false><<<gN, 256, 0, stream>>>(v_up + d * NC, Wprod_vT, mv_ + d * NC, N);
    // species-sorted sc matvec + state update + descriptor write
    dim3 gsc((N + 7) / 8, Z_DIM);
    sc_update<<<gsc, 64, 0, stream>>>(hist, perm, s_, v_, ms_, mv_,
                                      Wsc_s + (size_t)i * Z_DIM * 64 * 64,
                                      Wsc_v + (size_t)i * Z_DIM * 64 * 64,
                                      (float*)d_out, N, NC, i, L);
  }
}
